// AnchorEncoder_2903397892496
// MI455X (gfx1250) — matrix operand order repaired and checked by reading, not compiled, not run on hardware
//
#include <hip/hip_runtime.h>
#include <hip/hip_bf16.h>
#include <stdint.h>

#define N_ROWS  32768
#define H_DIM   1024
#define C_CLS   1000
#define C_PAD   1024
#define OUT_DIM 1024
#define LDK     40      // LDS stride (bf16 elems) for 32-wide K tiles: 80B, conflict-free, 16B aligned

typedef __attribute__((ext_vector_type(16))) __bf16 v16bf;
typedef __attribute__((ext_vector_type(8)))  float  v8f;

union FragU { uint4 u[2]; v16bf v; };

static __device__ __forceinline__ v8f wmma_bf16(v16bf a, v16bf b, v8f c) {
  return __builtin_amdgcn_wmma_f32_16x16x32_bf16(false, a, false, b, (short)0, c, false, false);
}

// ---- CDNA5 async global->LDS copy (16B per lane), tracked by ASYNCcnt ----
static __device__ __forceinline__ void async_cp16(const __bf16* g, __bf16* l) {
  const unsigned lofs =
      (unsigned)(uintptr_t)(__attribute__((address_space(3))) __bf16*)l;  // LDS byte address
  asm volatile("global_load_async_to_lds_b128 %0, %1, off"
               :: "v"(lofs), "v"((unsigned long long)(uintptr_t)g)
               : "memory");
}
static __device__ __forceinline__ void wait_async0() {
  asm volatile("s_wait_asynccnt 0x0" ::: "memory");
}

// A fragment 16x32 (rows row0..row0+15, K k0..k0+31) from row-major bf16, leading dim ld.
// ISA layout: lane<16 holds K 0..7 (v0-3) and 16..23 (v4-7); lane>=16 holds K 8..15 / 24..31.
static __device__ __forceinline__ v16bf a_frag(const __bf16* base, int ld, int row0, int k0, int lane) {
  const int half = lane >> 4, l = lane & 15;
  const __bf16* p = base + (size_t)(row0 + l) * ld + k0 + half * 8;
  FragU f;
  f.u[0] = *(const uint4*)p;
  f.u[1] = *(const uint4*)(p + 16);
  return f.v;
}

// B fragment 32x16 (cols col0..col0+15, K k0..k0+31); storage: column c contiguous in K, leading dim ld.
// ISA layout: lane<16 holds K 0..7 and 16..23 for col=lane, lane>=16 holds K 8..15 / 24..31.
static __device__ __forceinline__ v16bf b_frag(const __bf16* base, int ld, int col0, int k0, int lane) {
  const int half = lane >> 4, l = lane & 15;
  const __bf16* p = base + (size_t)(col0 + l) * ld + k0 + half * 8;
  FragU f;
  f.u[0] = *(const uint4*)p;
  f.u[1] = *(const uint4*)(p + 16);
  return f.v;
}

static __device__ __forceinline__ void split2(float x, __bf16& h, __bf16& l) {
  h = (__bf16)x;
  l = (__bf16)(x - (float)h);
}

// Issue async loads for one 128x32 B tile (hi+lo) into LDS buffers (4 x b128 per thread).
static __device__ __forceinline__ void stage_b_async(
    const __bf16* __restrict__ bh_g, const __bf16* __restrict__ bl_g,
    __bf16* SbH, __bf16* SbL, int n_blk, int k0, int tid) {
  #pragma unroll
  for (int j = 0; j < 2; ++j) {
    const int c = tid + j * 256;          // 512 16B-chunks per matrix
    const int n = c >> 2, kc = (c & 3) * 8;
    const size_t gofs = (size_t)(n_blk + n) * H_DIM + k0 + kc;
    const int lofs = n * LDK + kc;
    async_cp16(bh_g + gofs, SbH + lofs);
    async_cp16(bl_g + gofs, SbL + lofs);
  }
}

// ---- K1: normalize anchors, split raw & scaled anchors into bf16 hi/lo (padded rows -> 0) ----
__global__ __launch_bounds__(256) void prep_anchors_kernel(
    const float* __restrict__ anchors, __bf16* __restrict__ as_hi, __bf16* __restrict__ as_lo,
    __bf16* __restrict__ ar_hi, __bf16* __restrict__ ar_lo) {
  const int c = blockIdx.x;
  const int tid = threadIdx.x;
  const size_t rb = (size_t)c * H_DIM;
  if (c >= C_CLS) {
    for (int i = tid; i < H_DIM; i += 256) {
      as_hi[rb + i] = (__bf16)0.f; as_lo[rb + i] = (__bf16)0.f;
      ar_hi[rb + i] = (__bf16)0.f; ar_lo[rb + i] = (__bf16)0.f;
    }
    return;
  }
  float vals[4];
  float s = 0.f;
  #pragma unroll
  for (int j = 0; j < 4; ++j) { vals[j] = anchors[rb + tid + j * 256]; s += vals[j] * vals[j]; }
  for (int m = 16; m >= 1; m >>= 1) s += __shfl_xor(s, m, 32);
  __shared__ float wsum[8];
  const int wave = tid >> 5;
  if ((tid & 31) == 0) wsum[wave] = s;
  __syncthreads();
  float tot = 0.f;
  #pragma unroll
  for (int w = 0; w < 8; ++w) tot += wsum[w];
  const float scale = 1.f / fmaxf(sqrtf(tot), 1e-8f);
  #pragma unroll
  for (int j = 0; j < 4; ++j) {
    const int i = tid + j * 256;
    __bf16 h, lo;
    split2(vals[j], h, lo);          ar_hi[rb + i] = h; ar_lo[rb + i] = lo;
    split2(vals[j] * scale, h, lo);  as_hi[rb + i] = h; as_lo[rb + i] = lo;
  }
}

// ---- K2: transpose + split W_out [2H][OUT] into w1t/w2t [OUT][H] bf16 hi/lo ----
__global__ __launch_bounds__(256) void prep_w_kernel(
    const float* __restrict__ W, __bf16* __restrict__ w1t_hi, __bf16* __restrict__ w1t_lo,
    __bf16* __restrict__ w2t_hi, __bf16* __restrict__ w2t_lo) {
  const int i = blockIdx.x * 256 + threadIdx.x;     // 0 .. 2*H*OUT-1
  const int k = i >> 10, n = i & 1023;
  __bf16 h, l;
  split2(W[(size_t)k * OUT_DIM + n], h, l);
  if (k < H_DIM) {
    w1t_hi[(size_t)n * H_DIM + k] = h;  w1t_lo[(size_t)n * H_DIM + k] = l;
  } else {
    const int kk = k - H_DIM;
    w2t_hi[(size_t)n * H_DIM + kk] = h; w2t_lo[(size_t)n * H_DIM + kk] = l;
  }
}

// ---- K3: AW = anchors_raw @ W1   (tiny: 1024x1024x1024, split-bf16, one 16x16 tile per wave) ----
__global__ __launch_bounds__(256) void aw_gemm_kernel(
    const __bf16* __restrict__ a_hi, const __bf16* __restrict__ a_lo,
    const __bf16* __restrict__ b_hi, const __bf16* __restrict__ b_lo,
    float* __restrict__ AW) {
  const int wave = threadIdx.x >> 5, lane = threadIdx.x & 31;
  const int tile = blockIdx.x * 8 + wave;
  const int m0 = (tile >> 6) * 16, n0 = (tile & 63) * 16;
  v8f acc = (v8f){0, 0, 0, 0, 0, 0, 0, 0};
  for (int k0 = 0; k0 < H_DIM; k0 += 32) {
    v16bf ah = a_frag(a_hi, H_DIM, m0, k0, lane);
    v16bf al = a_frag(a_lo, H_DIM, m0, k0, lane);
    v16bf bh = b_frag(b_hi, H_DIM, n0, k0, lane);
    v16bf bl = b_frag(b_lo, H_DIM, n0, k0, lane);
    acc = wmma_bf16(ah, bh, acc);
    acc = wmma_bf16(ah, bl, acc);
    acc = wmma_bf16(al, bh, acc);
  }
  const int half = lane >> 4, l = lane & 15;
  #pragma unroll
  for (int r = 0; r < 8; ++r)
    AW[(size_t)(m0 + r + half * 8) * OUT_DIM + (n0 + l)] = acc[r];
}

// ---- K4: zero argmax merge keys ----
__global__ __launch_bounds__(256) void zero_keys_kernel(unsigned long long* __restrict__ keys) {
  keys[blockIdx.x * 256 + threadIdx.x] = 0ull;
}

// ---- K5: sim = features @ anchors_scaled^T (split-bf16), per-row argmax merged via atomicMax ----
__global__ __launch_bounds__(256) void sim_gemm_kernel(
    const float* __restrict__ feat, const __bf16* __restrict__ bh_g,
    const __bf16* __restrict__ bl_g, unsigned long long* __restrict__ keys) {
  __shared__ __bf16 Sa_hi[128 * LDK];
  __shared__ __bf16 Sa_lo[128 * LDK];
  __shared__ __bf16 Sb_hi[2][128 * LDK];   // double-buffered, filled by async-to-LDS
  __shared__ __bf16 Sb_lo[2][128 * LDK];
  const int tid = threadIdx.x;
  const int wave = tid >> 5, lane = tid & 31;
  const int m_blk = blockIdx.x * 128;
  const int n_blk = blockIdx.y * 128;
  v8f acc[8];
  #pragma unroll
  for (int t = 0; t < 8; ++t) acc[t] = (v8f){0, 0, 0, 0, 0, 0, 0, 0};

  stage_b_async(bh_g, bl_g, Sb_hi[0], Sb_lo[0], n_blk, 0, tid);

  int buf = 0;
  for (int k0 = 0; k0 < H_DIM; k0 += 32, buf ^= 1) {
    __syncthreads();                                   // Sa / Sb[buf^1] free for reuse
    for (int i = tid; i < 128 * 32; i += 256) {        // stage A: f32 -> bf16 hi/lo
      const int r = i >> 5, k = i & 31;
      __bf16 h, lo;
      split2(feat[(size_t)(m_blk + r) * H_DIM + k0 + k], h, lo);
      Sa_hi[r * LDK + k] = h;
      Sa_lo[r * LDK + k] = lo;
    }
    wait_async0();                                     // B tile for this step landed
    __syncthreads();
    if (k0 + 32 < H_DIM)                               // prefetch next B tile (overlaps compute)
      stage_b_async(bh_g, bl_g, Sb_hi[buf ^ 1], Sb_lo[buf ^ 1], n_blk, k0 + 32, tid);

    v16bf ah = a_frag(Sa_hi, LDK, wave * 16, 0, lane);
    v16bf al = a_frag(Sa_lo, LDK, wave * 16, 0, lane);
    #pragma unroll
    for (int t = 0; t < 8; ++t) {
      v16bf bh = b_frag(Sb_hi[buf], LDK, t * 16, 0, lane);
      v16bf bl = b_frag(Sb_lo[buf], LDK, t * 16, 0, lane);
      acc[t] = wmma_bf16(ah, bh, acc[t]);
      acc[t] = wmma_bf16(ah, bl, acc[t]);
      acc[t] = wmma_bf16(al, bh, acc[t]);
    }
  }

  // per-row argmax over this block's 128 columns, then global merge
  const int half = lane >> 4, l = lane & 15;
  #pragma unroll
  for (int r = 0; r < 8; ++r) {
    float bv = -3.402823466e38f;
    int bc = 0x7FFFFFFF;
    #pragma unroll
    for (int t = 0; t < 8; ++t) {
      const int col = n_blk + t * 16 + l;
      const float v = acc[t][r];
      if (col < C_CLS && (v > bv || (v == bv && col < bc))) { bv = v; bc = col; }
    }
    for (int m = 8; m >= 1; m >>= 1) {                 // xor-reduce within each 16-lane half
      const float ov = __shfl_xor(bv, m, 32);
      const int   oc = __shfl_xor(bc, m, 32);
      if (ov > bv || (ov == bv && oc < bc)) { bv = ov; bc = oc; }
    }
    if (l == 0) {
      const int row = m_blk + wave * 16 + half * 8 + r;
      const unsigned u = __float_as_uint(bv);
      const unsigned key32 = (u & 0x80000000u) ? ~u : (u | 0x80000000u);   // monotonic float map
      const unsigned long long key = ((unsigned long long)key32 << 32) | (unsigned)(~(unsigned)bc);
      atomicMax(&keys[row], key);
    }
  }
}

// ---- K6: unpack winning index ----
__global__ __launch_bounds__(256) void finalize_idx_kernel(
    const unsigned long long* __restrict__ keys, int* __restrict__ idx) {
  const int n = blockIdx.x * 256 + threadIdx.x;
  idx[n] = (int)(~(unsigned)(keys[n] & 0xFFFFFFFFull));
}

// ---- K7: out = features @ W2 (split-bf16) + AW[idx] gather-add ----
__global__ __launch_bounds__(256) void out_gemm_kernel(
    const float* __restrict__ feat, const __bf16* __restrict__ bh_g,
    const __bf16* __restrict__ bl_g, const float* __restrict__ AW,
    const int* __restrict__ idx, float* __restrict__ out) {
  __shared__ __bf16 Sa_hi[128 * LDK];
  __shared__ __bf16 Sa_lo[128 * LDK];
  __shared__ __bf16 Sb_hi[2][128 * LDK];
  __shared__ __bf16 Sb_lo[2][128 * LDK];
  const int tid = threadIdx.x;
  const int wave = tid >> 5, lane = tid & 31;
  const int m_blk = blockIdx.x * 128;
  const int n_blk = blockIdx.y * 128;
  v8f acc[8];
  #pragma unroll
  for (int t = 0; t < 8; ++t) acc[t] = (v8f){0, 0, 0, 0, 0, 0, 0, 0};

  stage_b_async(bh_g, bl_g, Sb_hi[0], Sb_lo[0], n_blk, 0, tid);

  int buf = 0;
  for (int k0 = 0; k0 < H_DIM; k0 += 32, buf ^= 1) {
    __syncthreads();
    for (int i = tid; i < 128 * 32; i += 256) {
      const int r = i >> 5, k = i & 31;
      __bf16 h, lo;
      split2(feat[(size_t)(m_blk + r) * H_DIM + k0 + k], h, lo);
      Sa_hi[r * LDK + k] = h;
      Sa_lo[r * LDK + k] = lo;
    }
    wait_async0();
    __syncthreads();
    if (k0 + 32 < H_DIM)
      stage_b_async(bh_g, bl_g, Sb_hi[buf ^ 1], Sb_lo[buf ^ 1], n_blk, k0 + 32, tid);

    v16bf ah = a_frag(Sa_hi, LDK, wave * 16, 0, lane);
    v16bf al = a_frag(Sa_lo, LDK, wave * 16, 0, lane);
    #pragma unroll
    for (int t = 0; t < 8; ++t) {
      v16bf bh = b_frag(Sb_hi[buf], LDK, t * 16, 0, lane);
      v16bf bl = b_frag(Sb_lo[buf], LDK, t * 16, 0, lane);
      acc[t] = wmma_bf16(ah, bh, acc[t]);
      acc[t] = wmma_bf16(ah, bl, acc[t]);
      acc[t] = wmma_bf16(al, bh, acc[t]);
    }
  }

  const int half = lane >> 4, l = lane & 15;
  #pragma unroll
  for (int r = 0; r < 8; ++r) {
    const int row = m_blk + wave * 16 + half * 8 + r;
    const float* awrow = AW + (size_t)idx[row] * OUT_DIM;
    #pragma unroll
    for (int t = 0; t < 8; ++t) {
      const int col = n_blk + t * 16 + l;
      out[(size_t)row * OUT_DIM + col] = acc[t][r] + awrow[col];
    }
  }
}

extern "C" void kernel_launch(void* const* d_in, const int* in_sizes, int n_in,
                              void* d_out, int out_size, void* d_ws, size_t ws_size,
                              hipStream_t stream) {
  const float* feat    = (const float*)d_in[0];
  const float* anchors = (const float*)d_in[1];
  const float* W       = (const float*)d_in[2];
  float* out = (float*)d_out;

  char* p = (char*)d_ws;
  const size_t SZ_BF = (size_t)C_PAD * H_DIM * sizeof(__bf16);   // 2 MB each
  __bf16* as_hi  = (__bf16*)p; p += SZ_BF;
  __bf16* as_lo  = (__bf16*)p; p += SZ_BF;
  __bf16* ar_hi  = (__bf16*)p; p += SZ_BF;
  __bf16* ar_lo  = (__bf16*)p; p += SZ_BF;
  __bf16* w1t_hi = (__bf16*)p; p += SZ_BF;
  __bf16* w1t_lo = (__bf16*)p; p += SZ_BF;
  __bf16* w2t_hi = (__bf16*)p; p += SZ_BF;
  __bf16* w2t_lo = (__bf16*)p; p += SZ_BF;
  float* AW = (float*)p;                     p += (size_t)C_PAD * OUT_DIM * sizeof(float);
  unsigned long long* keys = (unsigned long long*)p; p += (size_t)N_ROWS * sizeof(unsigned long long);
  int* idx = (int*)p;                        p += (size_t)N_ROWS * sizeof(int);
  // total ws use ~20.4 MB

  prep_anchors_kernel<<<C_PAD, 256, 0, stream>>>(anchors, as_hi, as_lo, ar_hi, ar_lo);
  prep_w_kernel<<<(2 * H_DIM * OUT_DIM) / 256, 256, 0, stream>>>(W, w1t_hi, w1t_lo, w2t_hi, w2t_lo);
  aw_gemm_kernel<<<(C_PAD / 16) * (OUT_DIM / 16) / 8, 256, 0, stream>>>(ar_hi, ar_lo, w1t_hi, w1t_lo, AW);
  zero_keys_kernel<<<N_ROWS / 256, 256, 0, stream>>>(keys);
  sim_gemm_kernel<<<dim3(N_ROWS / 128, C_PAD / 128), 256, 0, stream>>>(feat, as_hi, as_lo, keys);
  finalize_idx_kernel<<<N_ROWS / 256, 256, 0, stream>>>(keys, idx);
  out_gemm_kernel<<<dim3(N_ROWS / 128, OUT_DIM / 128), 256, 0, stream>>>(feat, w2t_hi, w2t_lo, AW, idx, out);
}
